// droneDeliveryModelHeterogenous_9723805958640
// MI455X (gfx1250) — matrix operand order repaired and checked by reading, not compiled, not run on hardware
//
#include <hip/hip_runtime.h>
#include <hip/hip_bf16.h>

typedef __attribute__((ext_vector_type(16))) __bf16 v16bf;
typedef __attribute__((ext_vector_type(8)))  __bf16 v8bf;
typedef __attribute__((ext_vector_type(8)))  float  v8f;

#define HEADS 8
#define CH    64
#define HC    512   // HEADS*CH
#define NEG_SLOPE 0.2f

// ---------------------------------------------------------------------------
// float atomic max via sign-split int/uint trick (order-preserving for IEEE)
// ---------------------------------------------------------------------------
__device__ __forceinline__ void atomicMaxF(float* addr, float v) {
  if (v >= 0.0f) atomicMax((int*)addr, __float_as_int(v));
  else           atomicMin((unsigned int*)addr, __float_as_uint(v));
}

// ---------------------------------------------------------------------------
// f32 -> bf16 conversion
// ---------------------------------------------------------------------------
__global__ void f2bf_kernel(const float* __restrict__ in, __bf16* __restrict__ out, int n) {
  int i = blockIdx.x * blockDim.x + threadIdx.x;
  if (i < n) out[i] = (__bf16)in[i];
}

// ---------------------------------------------------------------------------
// Input transform: per-node type-selected 6->64 linear, output bf16
// ---------------------------------------------------------------------------
__global__ void input_transform_kernel(const float* __restrict__ x,
                                       const float* __restrict__ W1d, const float* __restrict__ b1d,
                                       const float* __restrict__ W1g, const float* __restrict__ b1g,
                                       __bf16* __restrict__ h0, int Nn) {
  int i = blockIdx.x * blockDim.x + threadIdx.x;
  if (i >= Nn * CH) return;
  int n = i >> 6, j = i & 63;
  bool goal = x[n * 6 + 5] > 0.0f;
  const float* W = goal ? W1g : W1d;
  const float* b = goal ? b1g : b1d;
  float v = b[j];
  #pragma unroll
  for (int k = 0; k < 6; ++k) v += x[n * 6 + k] * W[j * 6 + k];
  h0[i] = (__bf16)v;
}

// ---------------------------------------------------------------------------
// Append self-loops to the edge list
// ---------------------------------------------------------------------------
__global__ void build_edges_kernel(const int* __restrict__ ei, int* __restrict__ s,
                                   int* __restrict__ d, int E, int Nn) {
  int i = blockIdx.x * blockDim.x + threadIdx.x;
  if (i < E)            { s[i] = ei[i]; d[i] = ei[E + i]; }
  else if (i < E + Nn)  { s[i] = i - E; d[i] = i - E; }
}

// ---------------------------------------------------------------------------
// WMMA GEMM: C[M,Nout] = A[M,64](bf16) x W[Nout,64]^T(bf16) + bias, f32 out
// One wave per 16x16 tile, K=64 -> 2x v_wmma_f32_16x16x32_bf16
// ---------------------------------------------------------------------------
__global__ void gemm_bf16_wmma_kernel(const __bf16* __restrict__ A,
                                      const __bf16* __restrict__ W,
                                      const float* __restrict__ bias,
                                      float* __restrict__ C, int M, int Nout) {
  int wave = (blockIdx.x * blockDim.x + threadIdx.x) >> 5;
  int lane = threadIdx.x & 31;
  int ntiles = Nout >> 4;
  int mt = wave / ntiles, nt = wave % ntiles;
  if (mt * 16 >= M) return;

  int row = mt * 16 + (lane & 15); if (row >= M) row = M - 1;
  int col = nt * 16 + (lane & 15);
  int kbA = (lane >> 4) << 3;   // 0 or 8  (A: K halves 0-7 / 8-15 within each 16)
  int kbB = (lane >> 4) << 3;   // 0 or 8  (B: same K order as A)

  const __bf16* arow = A + (size_t)row * 64;
  const __bf16* wrow = W + (size_t)col * 64;

  v8f acc = {};
  #pragma unroll
  for (int ko = 0; ko < 64; ko += 32) {
    v8bf alo = *(const v8bf*)(arow + ko + kbA);        // K = ko+kbA .. +7
    v8bf ahi = *(const v8bf*)(arow + ko + 16 + kbA);   // K = ko+16+kbA .. +7
    v16bf a = __builtin_shufflevector(alo, ahi, 0,1,2,3,4,5,6,7,8,9,10,11,12,13,14,15);
    v8bf blo = *(const v8bf*)(wrow + ko + kbB);        // K = ko+kbB .. +7
    v8bf bhi = *(const v8bf*)(wrow + ko + 16 + kbB);   // K = ko+16+kbB .. +7
    v16bf b = __builtin_shufflevector(blo, bhi, 0,1,2,3,4,5,6,7,8,9,10,11,12,13,14,15);
#if __HIP_DEVICE_COMPILE__
    acc = __builtin_amdgcn_wmma_f32_16x16x32_bf16(false, a, false, b, (short)0, acc, false, false);
#else
    (void)a; (void)b;
#endif
  }

  float bv = bias ? bias[col] : 0.0f;
  int baseRow = mt * 16 + ((lane >> 4) << 3);
  float* cp = C + (size_t)baseRow * Nout + nt * 16 + (lane & 15);
  if (mt * 16 + 16 <= M) {             // wave-uniform full-tile fast path
    #pragma unroll
    for (int r = 0; r < 8; ++r) cp[(size_t)r * Nout] = acc[r] + bv;
  } else {
    #pragma unroll
    for (int r = 0; r < 8; ++r)
      if (baseRow + r < M) cp[(size_t)r * Nout] = acc[r] + bv;
  }
}

// ---------------------------------------------------------------------------
// WMMA GEMM with per-M-tile weight selection (drone/goal), bf16 out, Nout=64
// halfM is a multiple of 16 by construction (N/2 = 625*16)
// ---------------------------------------------------------------------------
__global__ void gemm_dual_bf16_wmma_kernel(const __bf16* __restrict__ A,
                                           const __bf16* __restrict__ Wd,
                                           const __bf16* __restrict__ Wg,
                                           const float* __restrict__ bd,
                                           const float* __restrict__ bg,
                                           __bf16* __restrict__ C, int M, int halfM) {
  int wave = (blockIdx.x * blockDim.x + threadIdx.x) >> 5;
  int lane = threadIdx.x & 31;
  const int ntiles = CH >> 4;  // 4
  int mt = wave / ntiles, nt = wave % ntiles;
  if (mt * 16 >= M) return;

  bool goal = (mt * 16) >= halfM;
  const __bf16* W = goal ? Wg : Wd;
  const float*  b = goal ? bg : bd;

  int row = mt * 16 + (lane & 15); if (row >= M) row = M - 1;
  int col = nt * 16 + (lane & 15);
  int kbA = (lane >> 4) << 3;
  int kbB = (lane >> 4) << 3;

  const __bf16* arow = A + (size_t)row * 64;
  const __bf16* wrow = W + (size_t)col * 64;

  v8f acc = {};
  #pragma unroll
  for (int ko = 0; ko < 64; ko += 32) {
    v8bf alo = *(const v8bf*)(arow + ko + kbA);
    v8bf ahi = *(const v8bf*)(arow + ko + 16 + kbA);
    v16bf av = __builtin_shufflevector(alo, ahi, 0,1,2,3,4,5,6,7,8,9,10,11,12,13,14,15);
    v8bf blo = *(const v8bf*)(wrow + ko + kbB);
    v8bf bhi = *(const v8bf*)(wrow + ko + 16 + kbB);
    v16bf bv = __builtin_shufflevector(blo, bhi, 0,1,2,3,4,5,6,7,8,9,10,11,12,13,14,15);
#if __HIP_DEVICE_COMPILE__
    acc = __builtin_amdgcn_wmma_f32_16x16x32_bf16(false, av, false, bv, (short)0, acc, false, false);
#else
    (void)av; (void)bv;
#endif
  }

  float bias = b[col];
  int baseRow = mt * 16 + ((lane >> 4) << 3);
  __bf16* cp = C + (size_t)baseRow * CH + nt * 16 + (lane & 15);
  if (mt * 16 + 16 <= M) {
    #pragma unroll
    for (int r = 0; r < 8; ++r) cp[(size_t)r * CH] = (__bf16)(acc[r] + bias);
  } else {
    #pragma unroll
    for (int r = 0; r < 8; ++r)
      if (baseRow + r < M) cp[(size_t)r * CH] = (__bf16)(acc[r] + bias);
  }
}

// ---------------------------------------------------------------------------
// Init per-dst max / denominator
// ---------------------------------------------------------------------------
__global__ void init_md_kernel(float* __restrict__ m, float* __restrict__ den, int Nn) {
  int i = blockIdx.x * blockDim.x + threadIdx.x;
  if (i < Nn * HEADS) { m[i] = -3.0e38f; den[i] = 0.0f; }
}

__global__ void zero_f_kernel(float* __restrict__ p, int n) {
  int i = blockIdx.x * blockDim.x + threadIdx.x;
  if (i < n) p[i] = 0.0f;
}

// ---------------------------------------------------------------------------
// Pass 1: attention scores + segment max. One WAVE per edge: lane l covers
// channels [l*16, l*16+16) of the 512-float row -> the wave's 4 float4 loads
// together cover the contiguous 2048B row. Head h lives in lanes 4h..4h+3;
// 4-lane shfl_xor reduction yields the per-head score.
// ---------------------------------------------------------------------------
__global__ void edge_score_kernel(const float* __restrict__ xl, const float* __restrict__ xr,
                                  const int* __restrict__ s, const int* __restrict__ d,
                                  const float* __restrict__ att,
                                  float* __restrict__ score, float* __restrict__ m, int EN) {
  int wave = (blockIdx.x * blockDim.x + threadIdx.x) >> 5;
  int lane = threadIdx.x & 31;
  if (wave >= EN) return;
  int si = s[wave], di = d[wave];
  const float4* pl = (const float4*)(xl + (size_t)si * HC) + lane * 4;
  const float4* pr = (const float4*)(xr + (size_t)di * HC) + lane * 4;
  const float4* pa = (const float4*)att + lane * 4;
  float sc = 0.0f;
  #pragma unroll
  for (int c = 0; c < 4; ++c) {
    float4 a = pl[c], b = pr[c], w = pa[c];
    float v;
    v = a.x + b.x; v = v > 0.0f ? v : NEG_SLOPE * v; sc += v * w.x;
    v = a.y + b.y; v = v > 0.0f ? v : NEG_SLOPE * v; sc += v * w.y;
    v = a.z + b.z; v = v > 0.0f ? v : NEG_SLOPE * v; sc += v * w.z;
    v = a.w + b.w; v = v > 0.0f ? v : NEG_SLOPE * v; sc += v * w.w;
  }
  // reduce over the 4 lanes of each head group
  sc += __shfl_xor(sc, 1, 32);
  sc += __shfl_xor(sc, 2, 32);
  if ((lane & 3) == 0) {
    int h = lane >> 2;
    score[(size_t)wave * HEADS + h] = sc;
    atomicMaxF(&m[di * HEADS + h], sc);
  }
}

// ---------------------------------------------------------------------------
// Pass 2: exp(score - max) + segment sum (denominator)
// ---------------------------------------------------------------------------
__global__ void edge_softmax_kernel(float* __restrict__ score, const int* __restrict__ d,
                                    const float* __restrict__ m, float* __restrict__ den, int EN) {
  int t = blockIdx.x * blockDim.x + threadIdx.x;
  if (t >= EN * HEADS) return;
  int e = t >> 3, h = t & 7;
  int di = d[e];
  float num = __expf(score[t] - m[di * HEADS + h]);
  score[t] = num;
  atomicAdd(&den[di * HEADS + h], num);
}

// ---------------------------------------------------------------------------
// Pass 3: weighted scatter-add of source features. One WAVE per edge,
// lane covers 16 contiguous channels -> coalesced loads and atomics.
// ---------------------------------------------------------------------------
__global__ void edge_aggregate_kernel(const float* __restrict__ xl,
                                      const int* __restrict__ s, const int* __restrict__ d,
                                      const float* __restrict__ score, const float* __restrict__ den,
                                      float* __restrict__ agg, int EN) {
  int wave = (blockIdx.x * blockDim.x + threadIdx.x) >> 5;
  int lane = threadIdx.x & 31;
  if (wave >= EN) return;
  int si = s[wave], di = d[wave];
  int h = lane >> 2;
  float alpha = score[(size_t)wave * HEADS + h] / den[di * HEADS + h];
  const float4* pl = (const float4*)(xl + (size_t)si * HC) + lane * 4;
  float* po = agg + (size_t)di * HC + lane * 16;
  #pragma unroll
  for (int c = 0; c < 4; ++c) {
    float4 a = pl[c];
    atomicAdd(po + c * 4 + 0, a.x * alpha);
    atomicAdd(po + c * 4 + 1, a.y * alpha);
    atomicAdd(po + c * 4 + 2, a.z * alpha);
    atomicAdd(po + c * 4 + 3, a.w * alpha);
  }
}

// ---------------------------------------------------------------------------
// Mean over heads + bias + ReLU; optional f32 and/or bf16 outputs
// ---------------------------------------------------------------------------
__global__ void head_mean_kernel(const float* __restrict__ agg, const float* __restrict__ bias,
                                 float* __restrict__ outF, __bf16* __restrict__ outB, int Nn) {
  int i = blockIdx.x * blockDim.x + threadIdx.x;
  if (i >= Nn * CH) return;
  int n = i >> 6, c = i & 63;
  float sum = 0.0f;
  #pragma unroll
  for (int h = 0; h < HEADS; ++h) sum += agg[(size_t)n * HC + h * CH + c];
  float v = sum * (1.0f / HEADS) + bias[c];
  v = v > 0.0f ? v : 0.0f;
  if (outF) outF[i] = v;
  if (outB) outB[i] = (__bf16)v;
}

// ---------------------------------------------------------------------------
// Final linear 64 -> 5, only drone (first half) rows
// ---------------------------------------------------------------------------
__global__ void final_linear_kernel(const float* __restrict__ h, const float* __restrict__ Wo,
                                    const float* __restrict__ bo, float* __restrict__ y, int half) {
  int t = blockIdx.x * blockDim.x + threadIdx.x;
  if (t >= half * 5) return;
  int n = t / 5, o = t % 5;
  float v = bo[o];
  const float4* ph = (const float4*)(h + (size_t)n * CH);
  const float4* pw = (const float4*)(Wo + (size_t)o * CH);
  #pragma unroll
  for (int c = 0; c < CH / 4; ++c) {
    float4 a = ph[c], w = pw[c];
    v += a.x * w.x + a.y * w.y + a.z * w.z + a.w * w.w;
  }
  y[t] = v;
}

// ---------------------------------------------------------------------------
#define CEILDIV(a, b) (((a) + (b) - 1) / (b))

extern "C" void kernel_launch(void* const* d_in, const int* in_sizes, int n_in,
                              void* d_out, int out_size, void* d_ws, size_t ws_size,
                              hipStream_t stream) {
  const float* x     = (const float*)d_in[0];
  const int*   ei    = (const int*)  d_in[1];
  const float* W1d   = (const float*)d_in[2];
  const float* b1d   = (const float*)d_in[3];
  const float* W1g   = (const float*)d_in[4];
  const float* b1g   = (const float*)d_in[5];
  const float* Wl1   = (const float*)d_in[6];
  const float* bl1   = (const float*)d_in[7];
  const float* Wr1   = (const float*)d_in[8];
  const float* br1   = (const float*)d_in[9];
  const float* att1  = (const float*)d_in[10];
  const float* bias1 = (const float*)d_in[11];
  const float* W2d   = (const float*)d_in[12];
  const float* b2d   = (const float*)d_in[13];
  const float* W2g   = (const float*)d_in[14];
  const float* b2g   = (const float*)d_in[15];
  const float* Wl2   = (const float*)d_in[16];
  const float* bl2   = (const float*)d_in[17];
  const float* Wr2   = (const float*)d_in[18];
  const float* br2   = (const float*)d_in[19];
  const float* att2  = (const float*)d_in[20];
  const float* bias2 = (const float*)d_in[21];
  const float* Wo    = (const float*)d_in[22];
  const float* bo    = (const float*)d_in[23];

  const int N  = in_sizes[0] / 6;
  const int E  = in_sizes[1] / 2;
  const int EN = E + N;
  const int halfN = N / 2;

  // ---- workspace bump allocator -------------------------------------------
  char* wsb = (char*)d_ws;
  size_t off = 0;
  auto alloc = [&](size_t bytes) -> char* {
    char* p = wsb + off;
    off = (off + bytes + 255) & ~(size_t)255;
    return p;
  };

  __bf16* h0b   = (__bf16*)alloc((size_t)N * CH * 2);
  __bf16* Wl1b  = (__bf16*)alloc((size_t)HC * CH * 2);
  __bf16* Wr1b  = (__bf16*)alloc((size_t)HC * CH * 2);
  __bf16* Wl2b  = (__bf16*)alloc((size_t)HC * CH * 2);
  __bf16* Wr2b  = (__bf16*)alloc((size_t)HC * CH * 2);
  __bf16* W2db  = (__bf16*)alloc((size_t)CH * CH * 2);
  __bf16* W2gb  = (__bf16*)alloc((size_t)CH * CH * 2);
  __bf16* h1b   = (__bf16*)alloc((size_t)N * CH * 2);
  __bf16* h2pb  = (__bf16*)alloc((size_t)N * CH * 2);
  float*  h2f   = (float*) alloc((size_t)N * CH * 4);
  float*  xl    = (float*) alloc((size_t)N * HC * 4);
  float*  xr    = (float*) alloc((size_t)N * HC * 4);  // doubles as agg after score pass
  int*    sA    = (int*)   alloc((size_t)EN * 4);
  int*    dA    = (int*)   alloc((size_t)EN * 4);
  float*  score = (float*) alloc((size_t)EN * HEADS * 4);
  float*  mB    = (float*) alloc((size_t)N * HEADS * 4);
  float*  denB  = (float*) alloc((size_t)N * HEADS * 4);
  float*  agg   = xr;  // alias: xr dead after score pass

  const int B = 256;
  const int gemmWavesBig  = CEILDIV(N, 16) * (HC / 16);  // 40000 waves
  const int gemmWavesSmall= CEILDIV(N, 16) * (CH / 16);  //  5000 waves
  const int edgeBlocks    = CEILDIV(EN * 32, B);         // wave-per-edge kernels

  // ---- weight conversions --------------------------------------------------
  f2bf_kernel<<<CEILDIV(HC * CH, B), B, 0, stream>>>(Wl1, Wl1b, HC * CH);
  f2bf_kernel<<<CEILDIV(HC * CH, B), B, 0, stream>>>(Wr1, Wr1b, HC * CH);
  f2bf_kernel<<<CEILDIV(HC * CH, B), B, 0, stream>>>(Wl2, Wl2b, HC * CH);
  f2bf_kernel<<<CEILDIV(HC * CH, B), B, 0, stream>>>(Wr2, Wr2b, HC * CH);
  f2bf_kernel<<<CEILDIV(CH * CH, B), B, 0, stream>>>(W2d, W2db, CH * CH);
  f2bf_kernel<<<CEILDIV(CH * CH, B), B, 0, stream>>>(W2g, W2gb, CH * CH);

  // ---- input transform + edge list with self loops ------------------------
  input_transform_kernel<<<CEILDIV(N * CH, B), B, 0, stream>>>(x, W1d, b1d, W1g, b1g, h0b, N);
  build_edges_kernel<<<CEILDIV(EN, B), B, 0, stream>>>(ei, sA, dA, E, N);

  // ======================= Layer 1 =========================================
  gemm_bf16_wmma_kernel<<<CEILDIV(gemmWavesBig, 8), B, 0, stream>>>(h0b, Wl1b, bl1, xl, N, HC);
  gemm_bf16_wmma_kernel<<<CEILDIV(gemmWavesBig, 8), B, 0, stream>>>(h0b, Wr1b, br1, xr, N, HC);
  init_md_kernel<<<CEILDIV(N * HEADS, B), B, 0, stream>>>(mB, denB, N);
  edge_score_kernel<<<edgeBlocks, B, 0, stream>>>(xl, xr, sA, dA, att1, score, mB, EN);
  edge_softmax_kernel<<<CEILDIV(EN * HEADS, B), B, 0, stream>>>(score, dA, mB, denB, EN);
  zero_f_kernel<<<CEILDIV(N * HC, B), B, 0, stream>>>(agg, N * HC);
  edge_aggregate_kernel<<<edgeBlocks, B, 0, stream>>>(xl, sA, dA, score, denB, agg, EN);
  head_mean_kernel<<<CEILDIV(N * CH, B), B, 0, stream>>>(agg, bias1, nullptr, h1b, N);

  // ======================= Layer 2 =========================================
  gemm_dual_bf16_wmma_kernel<<<CEILDIV(gemmWavesSmall, 8), B, 0, stream>>>(
      h1b, W2db, W2gb, b2d, b2g, h2pb, N, halfN);
  gemm_bf16_wmma_kernel<<<CEILDIV(gemmWavesBig, 8), B, 0, stream>>>(h2pb, Wl2b, bl2, xl, N, HC);
  gemm_bf16_wmma_kernel<<<CEILDIV(gemmWavesBig, 8), B, 0, stream>>>(h2pb, Wr2b, br2, xr, N, HC);
  init_md_kernel<<<CEILDIV(N * HEADS, B), B, 0, stream>>>(mB, denB, N);
  edge_score_kernel<<<edgeBlocks, B, 0, stream>>>(xl, xr, sA, dA, att2, score, mB, EN);
  edge_softmax_kernel<<<CEILDIV(EN * HEADS, B), B, 0, stream>>>(score, dA, mB, denB, EN);
  zero_f_kernel<<<CEILDIV(N * HC, B), B, 0, stream>>>(agg, N * HC);
  edge_aggregate_kernel<<<edgeBlocks, B, 0, stream>>>(xl, sA, dA, score, denB, agg, EN);
  head_mean_kernel<<<CEILDIV(N * CH, B), B, 0, stream>>>(agg, bias2, h2f, nullptr, N);

  // ======================= Output head =====================================
  final_linear_kernel<<<CEILDIV(halfN * 5, B), B, 0, stream>>>(h2f, Wo, bo, (float*)d_out, halfN);

  (void)n_in; (void)out_size; (void)ws_size;
}
